// CDDModel_11519102287964
// MI455X (gfx1250) — matrix operand order repaired and checked by reading, not compiled, not run on hardware
//
// CDDModel (GVP-GNN) forward for MI455X (gfx1250), wave32 + WMMA bf16.
//
// Roofline: dominant cost = per-edge message GVPs (265x100 + 2x 116x100 GEMMs
// per edge per layer) and per-node FF GVPs -> ~45 GMAC (~90 GFLOP).
// Gathered-activation traffic ~1 GB -> ~45us at 23.3 TB/s HBM, so compute
// must run on the matrix pipes: every GEMM is v_wmma_f32_16x16x32_bf16 with
// fp32 accumulation on 16-row tiles staged in LDS. Weights are swizzled once
// per call (fp32 -> bf16 WMMA B-fragment order, zero padded) so the WMMA loop
// is pure 32B global loads (L2-resident, 192MB) + 16B LDS reads.
//
// Input layout assumption: harness flattens the setup_inputs() dict with
// jax-pytree ordering (nested dict keys sorted) -> 193 device pointers.

#include <hip/hip_runtime.h>
#include <cstdint>
#include <cstddef>

typedef __bf16 bf16_t;
typedef __attribute__((ext_vector_type(16))) __bf16 v16bf;
typedef __attribute__((ext_vector_type(8)))  float  v8f;

#define NS   100
#define NV   16
#define ESD  32
#define OUTD 512
#define NGR  64

__host__ __device__ static inline int pad16i(int x) { return (x + 15) & ~15; }
__host__ __device__ static inline int pad32i(int x) { return (x + 31) & ~31; }

// ---------------------------------------------------------------------------
// WMMA helpers (16x16x32 bf16, fp32 accumulate)
// ---------------------------------------------------------------------------
__device__ __forceinline__ v8f wmma_step(v16bf a, v16bf b, v8f c) {
  return __builtin_amdgcn_wmma_f32_16x16x32_bf16(false, a, false, b, (short)0, c,
                                                 false, false);
}

// A fragment from row-major LDS tile A[16][ldk] (bf16):
// lane l holds row m=l&15; elems 0..15 -> K = kb + (l>>4)*16 + e
// (contiguous order, matching the swizzled B fragments).
__device__ __forceinline__ v16bf ld_afrag(const bf16_t* A, int ldk, int kb, int lane) {
  const bf16_t* p = A + (lane & 15) * ldk + kb + ((lane >> 4) << 4);
  v16bf r;
#pragma unroll
  for (int e = 0; e < 8; ++e) { r[e] = p[e]; r[e + 8] = p[e + 8]; }
  return r;
}

// C = A(16xKp) * W(+bias). W is pre-swizzled B-fragment order:
// frag(ks,nt): 32 lanes x 16 contiguous bf16, lane n = l&15, K = (l>>4)*16+e.
__device__ __forceinline__ void gemm16(const bf16_t* A, int ldk, const bf16_t* Wf,
                                       const float* bias, int nreal, float* C,
                                       int ldn, int ksteps, int ntiles, int lane) {
  for (int nt = 0; nt < ntiles; ++nt) {
    v8f acc = {};
    for (int ks = 0; ks < ksteps; ++ks) {
      v16bf a = ld_afrag(A, ldk, ks << 5, lane);
      v16bf b = *(const v16bf*)(Wf + (((size_t)(ks * ntiles + nt) << 5) + lane) * 16);
      acc = wmma_step(a, b, acc);
    }
    int col = (nt << 4) + (lane & 15);
    int rb  = (lane >> 4) << 3;  // C rows: r + (lane>=16 ? 8 : 0)
    float bb = (bias != nullptr && col < nreal) ? bias[col] : 0.f;
#pragma unroll
    for (int r = 0; r < 8; ++r) C[(rb + r) * ldn + col] = acc[r] + bb;
  }
}

// ---------------------------------------------------------------------------
// Generic GVP on a 16-row tile (vector_gate=True, as in CDDModel)
// ---------------------------------------------------------------------------
struct GvpP {
  const bf16_t *wh, *ws, *wv, *wsv;  // swizzled bf16 fragments
  const float  *wsb, *wsvb;          // fp32 biases
  int si, vi, so, vo, h, relu;
};

// sbuf: [16][pad32(si+h)] bf16, cols 0..si-1 preset by caller (also used as
//       bf16 scratch for the gate GEMM A operand after ws is done).
// vbuf: [3][16][vcp] f32, channels 0..vi-1 preset; on exit holds gated vout,
//       channels >= vo zeroed.  vh: [3][16][hp] f32 scratch, hp = pad16(h).
// sout: [16][pad16(so)] f32 output (post-relu if relu).
// gate: [16][pad16(vo)] f32 scratch (may alias vh: vh is dead by then).
__device__ void gvp_tile(const GvpP P, int lane, bf16_t* sbuf, float* vbuf, int vcp,
                         bf16_t* tbuf, float* vh, int hp, float* sout, float* gate) {
  const int h   = P.h;
  const int sip = pad32i(P.si + h);
  const int sop = pad16i(P.so);
  if (P.vi > 0) {
    const int kp = pad32i(P.vi);
    for (int t = 0; t < 3; ++t) {  // vh = v @ wh, per coordinate
      for (int idx = lane; idx < 16 * kp; idx += 32) {
        int r = idx / kp, c = idx - r * kp;
        tbuf[idx] = (bf16_t)((c < P.vi) ? vbuf[(t * 16 + r) * vcp + c] : 0.f);
      }
      __syncthreads();
      gemm16(tbuf, kp, P.wh, nullptr, 0, vh + t * 16 * hp, hp, kp >> 5, hp >> 4, lane);
      __syncthreads();
    }
    const int w = sip - P.si;  // vn -> sbuf[:, si:sip], zero padded
    for (int idx = lane; idx < 16 * w; idx += 32) {
      int r = idx / w, c = idx - r * w;
      float val = 0.f;
      if (c < h) {
        float a = 0.f;
#pragma unroll
        for (int t = 0; t < 3; ++t) { float x = vh[(t * 16 + r) * hp + c]; a += x * x; }
        val = sqrtf(fmaxf(a, 1e-8f));
      }
      sbuf[r * sip + P.si + c] = (bf16_t)val;
    }
    __syncthreads();
  }
  gemm16(sbuf, sip, P.ws, P.wsb, P.so, sout, sop, sip >> 5, sop >> 4, lane);
  __syncthreads();
  if (P.vo > 0) {
    const int hk  = pad32i(h);
    const int vop = pad16i(P.vo);
    for (int t = 0; t < 3; ++t) {  // vout = vh @ wv  (written into vbuf)
      for (int idx = lane; idx < 16 * hk; idx += 32) {
        int r = idx / hk, c = idx - r * hk;
        tbuf[idx] = (bf16_t)((c < h) ? vh[(t * 16 + r) * hp + c] : 0.f);
      }
      __syncthreads();
      gemm16(tbuf, hk, P.wv, nullptr, 0, vbuf + t * 16 * vcp, vcp, hk >> 5, vop >> 4, lane);
      __syncthreads();
    }
    // gate = sigmoid(s_pre @ wsv + b); stage pre-relu s as bf16 in sbuf.
    const int sk = pad32i(P.so);
    for (int idx = lane; idx < 16 * sk; idx += 32) {
      int r = idx / sk, c = idx - r * sk;
      sbuf[idx] = (bf16_t)((c < P.so) ? sout[r * sop + c] : 0.f);
    }
    __syncthreads();
    gemm16(sbuf, sk, P.wsv, P.wsvb, P.vo, gate, vop, sk >> 5, vop >> 4, lane);
    __syncthreads();
    for (int idx = lane; idx < 3 * 16 * vcp; idx += 32) {
      int t = idx / (16 * vcp); int rem = idx - t * 16 * vcp;
      int r = rem / vcp, c = rem - r * vcp;
      float x = 0.f;
      if (c < P.vo) {
        float g = gate[r * vop + c];
        x = vbuf[idx] * (1.f / (1.f + __expf(-g)));
      }
      vbuf[idx] = x;
    }
    __syncthreads();
  }
  if (P.relu) {
    for (int idx = lane; idx < 16 * sop; idx += 32) sout[idx] = fmaxf(sout[idx], 0.f);
    __syncthreads();
  }
}

__device__ __forceinline__ void pack_s(const float* sout, int sop, int si, int sip,
                                       bf16_t* sbuf, int lane) {
  __syncthreads();
  for (int idx = lane; idx < 16 * si; idx += 32) {
    int r = idx / si, c = idx - r * si;
    sbuf[r * sip + c] = (bf16_t)sout[r * sop + c];
  }
  __syncthreads();
}

__device__ void ln_s16(float* sd, int C, const float* g, const float* b, int lane) {
  if (lane < 16) {
    float mu = 0.f;
    for (int c = 0; c < C; ++c) mu += sd[lane * C + c];
    mu /= (float)C;
    float var = 0.f;
    for (int c = 0; c < C; ++c) { float d = sd[lane * C + c] - mu; var += d * d; }
    var /= (float)C;
    float inv = rsqrtf(var + 1e-5f);
    for (int c = 0; c < C; ++c)
      sd[lane * C + c] = (sd[lane * C + c] - mu) * inv * g[c] + b[c];
  }
  __syncthreads();
}

// ---------------------------------------------------------------------------
// Kernels
// ---------------------------------------------------------------------------
__global__ void k_swizzle(const float* W, bf16_t* dst, int K, int N, int Kp, int Np) {
  int tid = blockIdx.x * blockDim.x + threadIdx.x;
  int ntiles = Np >> 4;
  int nfrag = (Kp >> 5) * ntiles;
  if (tid >= nfrag * 32) return;
  int frag = tid >> 5, lane = tid & 31;
  int ks = frag / ntiles, nt = frag - ks * ntiles;
  int n  = (nt << 4) + (lane & 15);
  int kb = (ks << 5) + ((lane >> 4) << 4);
  bf16_t* out = dst + (size_t)tid * 16;
#pragma unroll
  for (int e = 0; e < 16; ++e) {
    int k = kb + e;
    float x = (k < K && n < N) ? W[(size_t)k * N + n] : 0.f;
    out[e] = (bf16_t)x;
  }
}

__global__ void k_zero_f(float* p, long long n) {
  long long i = (long long)blockIdx.x * blockDim.x + threadIdx.x;
  if (i < n) p[i] = 0.f;
}

__global__ void k_embed(const int* atoms, const float* emb, const float* lng,
                        const float* lnb, const float* w, const float* b,
                        float* s, float* v, int N) {
  int n = blockIdx.x * blockDim.x + threadIdx.x;
  if (n >= N) return;
  float hs[13];
  int a = atoms[n];
  float mu = 0.f;
  for (int i = 0; i < 13; ++i) { hs[i] = emb[a * 13 + i]; mu += hs[i]; }
  mu /= 13.f;
  float var = 0.f;
  for (int i = 0; i < 13; ++i) { float d = hs[i] - mu; var += d * d; }
  var /= 13.f;
  float inv = rsqrtf(var + 1e-5f);
  for (int i = 0; i < 13; ++i) hs[i] = (hs[i] - mu) * inv * lng[i] + lnb[i];
  for (int o = 0; o < NS; ++o) {
    float acc = b[o];
    for (int i = 0; i < 13; ++i) acc += hs[i] * w[i * NS + o];
    s[(size_t)n * NS + o] = acc;
  }
  for (int c = 0; c < 48; ++c) v[(size_t)n * 48 + c] = 0.f;  // node vectors start 0
}

__global__ void k_edge_init(const float* edge_s, const float* edge_v,
                            const float* lng, const float* lnb, const float* wh,
                            const float* wsb, const float* wsw, const float* wsvb,
                            const float* wsvw, const float* wv, float* es, float* ev,
                            int E) {
  int e = blockIdx.x * blockDim.x + threadIdx.x;
  if (e >= E) return;
  float x[16];
  float mu = 0.f;
  for (int i = 0; i < 16; ++i) { x[i] = edge_s[(size_t)e * 16 + i]; mu += x[i]; }
  mu /= 16.f;
  float var = 0.f;
  for (int i = 0; i < 16; ++i) { float d = x[i] - mu; var += d * d; }
  var /= 16.f;
  float inv = rsqrtf(var + 1e-5f);
  for (int i = 0; i < 16; ++i) x[i] = (x[i] - mu) * inv * lng[i] + lnb[i];
  float v3[3];
  float n2 = 0.f;
  for (int t = 0; t < 3; ++t) { v3[t] = edge_v[(size_t)e * 3 + t]; n2 += v3[t] * v3[t]; }
  float ivn = rsqrtf(fmaxf(n2, 1e-8f));  // _ln_v with 1 channel
  float whv = wh[0], wvv = wv[0];
  float vh3[3];
  float h2 = 0.f;
  for (int t = 0; t < 3; ++t) { vh3[t] = v3[t] * ivn * whv; h2 += vh3[t] * vh3[t]; }
  float vn = sqrtf(fmaxf(h2, 1e-8f));
  float so[ESD];
  for (int o = 0; o < ESD; ++o) {
    float acc = wsb[o];
    for (int k = 0; k < 16; ++k) acc += x[k] * wsw[k * ESD + o];
    acc += vn * wsw[16 * ESD + o];
    so[o] = acc;
  }
  float gt = wsvb[0];
  for (int o = 0; o < ESD; ++o) gt += so[o] * wsvw[o];
  float sg = 1.f / (1.f + __expf(-gt));
  for (int t = 0; t < 3; ++t) ev[(size_t)e * 3 + t] = vh3[t] * wvv * sg;
  for (int o = 0; o < ESD; ++o) es[(size_t)e * ESD + o] = so[o];
}

__global__ void k_deg(const int* ei, float* deg, int E) {
  int e = blockIdx.x * blockDim.x + threadIdx.x;
  if (e < E) atomicAdd(&deg[ei[E + e]], 1.f);
}
__global__ void k_invdeg(const float* deg, float* invd, int N) {
  int n = blockIdx.x * blockDim.x + threadIdx.x;
  if (n < N) invd[n] = 1.f / fmaxf(deg[n], 1.f);
}

struct EdgeArgs {
  const float *s, *v, *es, *ev;
  const int* ei;
  float *agg_s, *agg_v;
  int E;
  GvpP m0, m1, m2;
};

__global__ __launch_bounds__(32) void k_edge_msg(EdgeArgs A) {
  __shared__ bf16_t sbuf[16 * 288];
  __shared__ float  vbuf[3 * 16 * 64];
  __shared__ bf16_t tbuf[16 * 64];
  __shared__ float  vh[3 * 16 * 48];
  __shared__ float  sout[16 * 112];
  __shared__ int    sidx[16], didx[16];
  int lane = threadIdx.x;
  int eb = blockIdx.x * 16;
  int rows = A.E - eb; if (rows > 16) rows = 16;
  if (lane < 16) {
    int e = eb + (lane < rows ? lane : 0);
    sidx[lane] = A.ei[e];
    didx[lane] = A.ei[A.E + e];
  }
  __syncthreads();
  // ms = [s_src(100) | es(32) | s_dst(100)], bf16, stride pad32(232+33)=288
  for (int idx = lane; idx < 16 * 232; idx += 32) {
    int r = idx / 232, c = idx - r * 232;
    int e = eb + (r < rows ? r : 0);
    float x;
    if (c < NS)            x = A.s[(size_t)sidx[r] * NS + c];
    else if (c < NS + ESD) x = A.es[(size_t)e * ESD + (c - NS)];
    else                   x = A.s[(size_t)didx[r] * NS + (c - NS - ESD)];
    sbuf[r * 288 + c] = (bf16_t)x;
  }
  // mv = [v_src(16) | ev(1) | v_dst(16)] per coord, zero pad to 64
  for (int idx = lane; idx < 3 * 16 * 64; idx += 32) {
    int t = idx / (16 * 64); int rem = idx - t * 16 * 64;
    int r = rem / 64, c = rem - r * 64;
    int e = eb + (r < rows ? r : 0);
    float x = 0.f;
    if (c < NV)              x = A.v[((size_t)sidx[r] * NV + c) * 3 + t];
    else if (c == NV)        x = A.ev[(size_t)e * 3 + t];
    else if (c < 2 * NV + 1) x = A.v[((size_t)didx[r] * NV + (c - NV - 1)) * 3 + t];
    vbuf[idx] = x;
  }
  __syncthreads();
  gvp_tile(A.m0, lane, sbuf, vbuf, 64, tbuf, vh, 48, sout, vh);
  pack_s(sout, 112, 100, 128, sbuf, lane);
  gvp_tile(A.m1, lane, sbuf, vbuf, 64, tbuf, vh, 16, sout, vh);
  pack_s(sout, 112, 100, 128, sbuf, lane);
  gvp_tile(A.m2, lane, sbuf, vbuf, 64, tbuf, vh, 16, sout, vh);
  for (int idx = lane; idx < rows * NS; idx += 32) {
    int r = idx / NS, c = idx - r * NS;
    atomicAdd(&A.agg_s[(size_t)didx[r] * NS + c], sout[r * 112 + c]);
  }
  for (int idx = lane; idx < rows * NV * 3; idx += 32) {
    int r = idx / (NV * 3); int rem = idx - r * NV * 3;
    int c = rem / 3, t = rem - c * 3;
    atomicAdd(&A.agg_v[((size_t)didx[r] * NV + c) * 3 + t], vbuf[(t * 16 + r) * 64 + c]);
  }
}

struct NodeArgs {
  float *s, *v;
  const float *agg_s, *agg_v, *invd;
  const float *ln0g, *ln0b, *ln1g, *ln1b;
  int N;
  GvpP ff0, ff1;
};

__global__ __launch_bounds__(32) void k_node(NodeArgs A) {
  __shared__ bf16_t sbuf[16 * 448];
  __shared__ float  vbuf[3 * 16 * 32];
  __shared__ bf16_t tbuf[16 * 32];
  __shared__ float  vh[3 * 16 * 32];
  __shared__ float  sout[16 * 400];
  __shared__ float  sres[16 * NS];
  __shared__ float  vres[16 * 48];
  __shared__ float  dnm[16];
  int lane = threadIdx.x;
  int nb = blockIdx.x * 16;
  int rows = A.N - nb; if (rows > 16) rows = 16;
  for (int idx = lane; idx < 16 * NS; idx += 32) {
    int r = idx / NS, c = idx - r * NS;
    size_t n = nb + (r < rows ? r : 0);
    sres[idx] = A.s[n * NS + c] + A.agg_s[n * NS + c] * A.invd[n];
  }
  for (int idx = lane; idx < 16 * 48; idx += 32) {
    int r = idx / 48, c = idx - r * 48;
    size_t n = nb + (r < rows ? r : 0);
    vres[idx] = A.v[n * 48 + c] + A.agg_v[n * 48 + c] * A.invd[n];
  }
  __syncthreads();
  ln_s16(sres, NS, A.ln0g, A.ln0b, lane);
  if (lane < 16) {  // _ln_v
    float acc = 0.f;
    for (int c = 0; c < NV; ++c) {
      float s2 = 0.f;
      for (int t = 0; t < 3; ++t) { float x = vres[lane * 48 + c * 3 + t]; s2 += x * x; }
      acc += fmaxf(s2, 1e-8f);
    }
    dnm[lane] = sqrtf(acc / (float)NV);
  }
  __syncthreads();
  for (int idx = lane; idx < 16 * 48; idx += 32) vres[idx] /= dnm[idx / 48];
  __syncthreads();
  for (int idx = lane; idx < 3 * 16 * 32; idx += 32) {
    int t = idx / (16 * 32); int rem = idx - t * 16 * 32;
    int r = rem / 32, c = rem - r * 32;
    vbuf[idx] = (c < NV) ? vres[r * 48 + c * 3 + t] : 0.f;
  }
  for (int idx = lane; idx < 16 * NS; idx += 32) {
    int r = idx / NS, c = idx - r * NS;
    sbuf[r * 160 + c] = (bf16_t)sres[idx];  // ff0: sip = pad32(100+32)=160
  }
  __syncthreads();
  gvp_tile(A.ff0, lane, sbuf, vbuf, 32, tbuf, vh, 32, sout, vh);
  pack_s(sout, 400, 400, 448, sbuf, lane);  // ff1: sip = pad32(400+32)=448
  gvp_tile(A.ff1, lane, sbuf, vbuf, 32, tbuf, vh, 32, sout, vh);
  for (int idx = lane; idx < 16 * NS; idx += 32) {
    int r = idx / NS, c = idx - r * NS;
    sres[idx] += sout[r * 112 + c];
  }
  __syncthreads();
  ln_s16(sres, NS, A.ln1g, A.ln1b, lane);
  for (int idx = lane; idx < 16 * 48; idx += 32) {
    int r = idx / 48; int rem = idx - r * 48;
    int c = rem / 3, t = rem - c * 3;
    vres[idx] += vbuf[(t * 16 + r) * 32 + c];
  }
  __syncthreads();
  if (lane < 16) {
    float acc = 0.f;
    for (int c = 0; c < NV; ++c) {
      float s2 = 0.f;
      for (int t = 0; t < 3; ++t) { float x = vres[lane * 48 + c * 3 + t]; s2 += x * x; }
      acc += fmaxf(s2, 1e-8f);
    }
    dnm[lane] = sqrtf(acc / (float)NV);
  }
  __syncthreads();
  for (int idx = lane; idx < rows * NS; idx += 32) {
    int r = idx / NS, c = idx - r * NS;
    A.s[(size_t)(nb + r) * NS + c] = sres[r * NS + c];
  }
  for (int idx = lane; idx < rows * 48; idx += 32) {
    int r = idx / 48, c = idx - r * 48;
    A.v[(size_t)(nb + r) * 48 + c] = vres[r * 48 + c] / dnm[r];
  }
}

struct OutArgs {
  const float *s, *v;
  const int* batch;
  const float *lng, *lnb;
  float* out;
  int N;
  GvpP wo;
};

__global__ __launch_bounds__(32) void k_out(OutArgs A) {
  __shared__ bf16_t sbuf[16 * 128];
  __shared__ float  vbuf[3 * 16 * 32];
  __shared__ bf16_t tbuf[16 * 32];
  __shared__ float  vh[3 * 16 * 16];
  __shared__ float  sout[16 * OUTD];
  __shared__ float  sres[16 * NS];
  __shared__ float  dnm[16];
  int lane = threadIdx.x;
  int nb = blockIdx.x * 16;
  int rows = A.N - nb; if (rows > 16) rows = 16;
  for (int idx = lane; idx < 16 * NS; idx += 32) {
    int r = idx / NS, c = idx - r * NS;
    size_t n = nb + (r < rows ? r : 0);
    sres[idx] = A.s[n * NS + c];
  }
  __syncthreads();
  ln_s16(sres, NS, A.lng, A.lnb, lane);
  if (lane < 16) {
    size_t n = nb + (lane < rows ? lane : 0);
    float acc = 0.f;
    for (int c = 0; c < NV; ++c) {
      float s2 = 0.f;
      for (int t = 0; t < 3; ++t) { float x = A.v[n * 48 + c * 3 + t]; s2 += x * x; }
      acc += fmaxf(s2, 1e-8f);
    }
    dnm[lane] = sqrtf(acc / (float)NV);
  }
  __syncthreads();
  for (int idx = lane; idx < 3 * 16 * 32; idx += 32) {
    int t = idx / (16 * 32); int rem = idx - t * 16 * 32;
    int r = rem / 32, c = rem - r * 32;
    size_t n = nb + (r < rows ? r : 0);
    vbuf[idx] = (c < NV) ? A.v[n * 48 + c * 3 + t] / dnm[r] : 0.f;
  }
  for (int idx = lane; idx < 16 * NS; idx += 32) {
    int r = idx / NS, c = idx - r * NS;
    sbuf[r * 128 + c] = (bf16_t)sres[idx];  // wout: sip = pad32(100+16)=128
  }
  __syncthreads();
  gvp_tile(A.wo, lane, sbuf, vbuf, 32, tbuf, vh, 16, sout, vh);
  for (int idx = lane; idx < rows * OUTD; idx += 32) {
    int r = idx / OUTD, c = idx - r * OUTD;
    atomicAdd(&A.out[(size_t)A.batch[nb + r] * OUTD + c], sout[r * OUTD + c]);
  }
}

__global__ void k_cnt(const int* batch, float* cnt, int N) {
  int n = blockIdx.x * blockDim.x + threadIdx.x;
  if (n < N) atomicAdd(&cnt[batch[n]], 1.f);
}
__global__ void k_div(float* out, const float* cnt, int n) {
  int i = blockIdx.x * blockDim.x + threadIdx.x;
  if (i < n) out[i] /= fmaxf(cnt[i / OUTD], 1.f);
}

// ---------------------------------------------------------------------------
// Host
// ---------------------------------------------------------------------------
extern "C" void kernel_launch(void* const* d_in, const int* in_sizes, int n_in,
                              void* d_out, int out_size, void* d_ws, size_t ws_size,
                              hipStream_t stream) {
  (void)n_in; (void)ws_size;
  const int N = in_sizes[0];
  const int E = in_sizes[3] / 2;
  const int*   atoms  = (const int*)d_in[0];
  const float* edge_s = (const float*)d_in[1];
  const float* edge_v = (const float*)d_in[2];
  const int*   ei     = (const int*)d_in[3];
  const int*   batch  = (const int*)d_in[4];
  float* out = (float*)d_out;

  char* ws = (char*)d_ws;
  size_t off = 0;
  auto alloc = [&](size_t bytes) -> void* {
    void* p = ws + off;
    off += (bytes + 255) & ~(size_t)255;
    return p;
  };
  float* s     = (float*)alloc((size_t)N * NS * sizeof(float));
  float* v     = (float*)alloc((size_t)N * 48 * sizeof(float));
  float* es    = (float*)alloc((size_t)E * ESD * sizeof(float));
  float* ev    = (float*)alloc((size_t)E * 3 * sizeof(float));
  float* agg   = (float*)alloc((size_t)N * 148 * sizeof(float));
  float* agg_s = agg;
  float* agg_v = agg + (size_t)N * NS;
  float* deg   = (float*)alloc((size_t)N * sizeof(float));
  float* invd  = (float*)alloc((size_t)N * sizeof(float));
  float* cnt   = (float*)alloc(NGR * sizeof(float));
  bf16_t* arena = (bf16_t*)(ws + off);  // bf16 weight-fragment arena (~2.3 MB)
  size_t aoff = 0;

  auto swz = [&](int idx, int K, int Nn) -> const bf16_t* {
    bf16_t* dst = arena + aoff;
    int Kp = pad32i(K), Np = pad16i(Nn);
    aoff += (size_t)Kp * Np;
    int threads = (Kp / 32) * (Np / 16) * 32;
    k_swizzle<<<(threads + 255) / 256, 256, 0, stream>>>((const float*)d_in[idx], dst,
                                                         K, Nn, Kp, Np);
    return dst;
  };
  // GVP leaf order within a param subtree (sorted keys): wh, ws.b, ws.w,
  // [wsv.b, wsv.w, wv] when vo > 0.
  auto mk = [&](int base, int si, int vi, int so, int vo, int relu) -> GvpP {
    GvpP p{};
    p.si = si; p.vi = vi; p.so = so; p.vo = vo; p.relu = relu;
    p.h = (vi > vo ? vi : vo);
    p.wh  = swz(base + 0, vi, p.h);
    p.wsb = (const float*)d_in[base + 1];
    p.ws  = swz(base + 2, si + p.h, so);
    if (vo) {
      p.wsvb = (const float*)d_in[base + 3];
      p.wsv  = swz(base + 4, so, vo);
      p.wv   = swz(base + 5, p.h, vo);
    }
    return p;
  };

  long long nz = (long long)N;
  k_zero_f<<<(unsigned)((nz + 255) / 256), 256, 0, stream>>>(deg, nz);
  k_zero_f<<<1, 256, 0, stream>>>(cnt, NGR);
  nz = out_size;
  k_zero_f<<<(unsigned)((nz + 255) / 256), 256, 0, stream>>>(out, nz);

  // d_in indices per jax-pytree flattening (see analysis): embed=5,
  // layer l base L=6+34l: ff0@L, ff1@L+6, ln0.b/g@L+12/13, ln1.b/g@L+14/15,
  // m0@L+16, m1@L+22, m2@L+28; we@176..181, we_ln b/g@182/183,
  // wo_ln b/g@184/185, wout@186..188, wv ws.b/w@189/190, wv_ln b/g@191/192.
  k_embed<<<(N + 127) / 128, 128, 0, stream>>>(
      atoms, (const float*)d_in[5], (const float*)d_in[192], (const float*)d_in[191],
      (const float*)d_in[190], (const float*)d_in[189], s, v, N);
  k_edge_init<<<(E + 127) / 128, 128, 0, stream>>>(
      edge_s, edge_v, (const float*)d_in[183], (const float*)d_in[182],
      (const float*)d_in[176], (const float*)d_in[177], (const float*)d_in[178],
      (const float*)d_in[179], (const float*)d_in[180], (const float*)d_in[181],
      es, ev, E);
  k_deg<<<(E + 255) / 256, 256, 0, stream>>>(ei, deg, E);
  k_invdeg<<<(N + 255) / 256, 256, 0, stream>>>(deg, invd, N);

  for (int l = 0; l < 5; ++l) {
    int L = 6 + 34 * l;
    GvpP f0 = mk(L + 0, 100, 16, 400, 32, 1);
    GvpP f1 = mk(L + 6, 400, 32, 100, 16, 0);
    GvpP m0 = mk(L + 16, 232, 33, 100, 16, 1);
    GvpP m1 = mk(L + 22, 100, 16, 100, 16, 1);
    GvpP m2 = mk(L + 28, 100, 16, 100, 16, 0);
    nz = (long long)N * 148;
    k_zero_f<<<(unsigned)((nz + 255) / 256), 256, 0, stream>>>(agg, nz);
    EdgeArgs ea;
    ea.s = s; ea.v = v; ea.es = es; ea.ev = ev; ea.ei = ei;
    ea.agg_s = agg_s; ea.agg_v = agg_v; ea.E = E;
    ea.m0 = m0; ea.m1 = m1; ea.m2 = m2;
    k_edge_msg<<<(E + 15) / 16, 32, 0, stream>>>(ea);
    NodeArgs na;
    na.s = s; na.v = v; na.agg_s = agg_s; na.agg_v = agg_v; na.invd = invd;
    na.ln0g = (const float*)d_in[L + 13]; na.ln0b = (const float*)d_in[L + 12];
    na.ln1g = (const float*)d_in[L + 15]; na.ln1b = (const float*)d_in[L + 14];
    na.N = N; na.ff0 = f0; na.ff1 = f1;
    k_node<<<(N + 15) / 16, 32, 0, stream>>>(na);
  }
  GvpP wo = mk(186, 100, 16, 512, 0, 1);
  OutArgs oa;
  oa.s = s; oa.v = v; oa.batch = batch;
  oa.lng = (const float*)d_in[185]; oa.lnb = (const float*)d_in[184];
  oa.out = out; oa.N = N; oa.wo = wo;
  k_out<<<(N + 15) / 16, 32, 0, stream>>>(oa);
  k_cnt<<<(N + 255) / 256, 256, 0, stream>>>(batch, cnt, N);
  k_div<<<(out_size + 255) / 256, 256, 0, stream>>>(out, cnt, out_size);
}
